// TransmorpherLayer2d_48893907698065
// MI455X (gfx1250) — matrix operand order repaired and checked by reading, not compiled, not run on hardware
//
#include <hip/hip_runtime.h>
#include <cstdint>

// ---------------- problem dims (fixed by reference) ----------------
#define DD   512
#define SS   64
#define LLEN 256
#define NN   16384      // S*L
#define FFD  2048
#define HH   64
#define EPSV 1e-5f

typedef __attribute__((ext_vector_type(16))) __bf16 v16bf;
typedef __attribute__((ext_vector_type(8)))  float  v8f;
typedef int v4i __attribute__((ext_vector_type(4)));
typedef int v8i __attribute__((ext_vector_type(8)));

// fragment = two 16-byte chunks, loaded as b128 each
union Frag128 { uint4 q[2]; v16bf v; };

__device__ __forceinline__ uint16_t f2bf(float f) {
  union { float f; uint32_t u; } c; c.f = f;
  uint32_t u = c.u;
  u += 0x7fffu + ((u >> 16) & 1u);   // round-to-nearest-even
  return (uint16_t)(u >> 16);
}

__device__ __forceinline__ v8f wmma_bf16(v16bf a, v16bf b, v8f c) {
  return __builtin_amdgcn_wmma_f32_16x16x32_bf16(false, a, false, b, (short)0, c, false, false);
}

// A fragment: lane holds row `sub`; VGPR v holds K = (v/4)*16 + half*8 + (v%4)*2
// -> elements [0..7] and [16..23] relative to p = base + half*8  (both 16B aligned)
__device__ __forceinline__ v16bf load_frag_a(const uint16_t* p) {
  Frag128 f;
  f.q[0] = *reinterpret_cast<const uint4*>(p);
  f.q[1] = *reinterpret_cast<const uint4*>(p + 16);
  return f.v;
}
// B fragment: lane holds col `sub`; VGPR v holds K = (v/4)*16 + half*8 + (v%4)*2
// -> elements [0..7] and [16..23] relative to p = base + half*8  (both 16B aligned)
__device__ __forceinline__ v16bf load_frag_b(const uint16_t* p) {
  Frag128 f;
  f.q[0] = *reinterpret_cast<const uint4*>(p);
  f.q[1] = *reinterpret_cast<const uint4*>(p + 16);
  return f.v;
}

// ---------------- TDM: issue one 2D tile load (128 rows x 256 bf16) to LDS ----------------
// D# group0/group1 per CDNA5 ISA.
// data_size=2B, pad_enable=1, pad_interval=128 DWORDs (code 6), pad_amount=4 DWORDs (code 3)
// -> LDS row stride = 512B + 16B = 528B (bank-conflict-free fragment reads).
#define KC     256
#define LROW   264                 // padded LDS row stride in elements
#define CHUNKE (128 * LROW)        // elements per staged tile (33792)
#define SMEM_BYTES (4 * CHUNKE * 2)  // A0,B0,A1,B1

__device__ __forceinline__ void tdm_issue(uint64_t gaddr, uint32_t lds_byte_off,
                                          uint32_t tdim0, uint32_t tdim1, uint32_t stride0) {
  v4i g0; v8i g1;
  g0[0] = 1;                                   // count=1, user mode
  g0[1] = (int)lds_byte_off;                   // lds_addr
  g0[2] = (int)(uint32_t)gaddr;                // global_addr[31:0]
  g0[3] = (int)(((uint32_t)(gaddr >> 32) & 0x01FFFFFFu) | (2u << 30)); // addr[56:32] | type=2
  g1[0] = (int)((1u << 16) | (1u << 20) | (6u << 22) | (3u << 25));    // dsz=2B, pad en/int/amt
  g1[1] = (int)((tdim0 & 0xFFFFu) << 16);                  // tensor_dim0[15:0]
  g1[2] = (int)((tdim0 >> 16) | ((tdim1 & 0xFFFFu) << 16)); // td0[31:16] | td1[15:0]
  g1[3] = (int)((tdim1 >> 16) | ((uint32_t)KC << 16));      // td1[31:16] | tile_dim0=256
  g1[4] = (int)(128u);                                      // tile_dim1=128, tile_dim2=0
  g1[5] = (int)stride0;                                     // tensor_dim0_stride[31:0]
  g1[6] = 0;
  g1[7] = 0;
  asm volatile("tensor_load_to_lds %0, %1" :: "s"(g0), "s"(g1) : "memory");
}

// ---------------- elementwise casts ----------------
__global__ __launch_bounds__(256)
void cast_f32_bf16(const float* __restrict__ src, uint16_t* __restrict__ dst, int n) {
  int i = blockIdx.x * 256 + threadIdx.x;
  if (i < n) dst[i] = f2bf(src[i]);
}

// xT[n][d] = bf16(x[d][n]); X is [512][16384]
__global__ __launch_bounds__(256)
void transpose_cast_x(const float* __restrict__ X, uint16_t* __restrict__ XT) {
  __shared__ float tile[32][33];
  int c0 = blockIdx.x * 32;
  int r0 = blockIdx.y * 32;
  int tx = threadIdx.x & 31;
  int ty = threadIdx.x >> 5;
  #pragma unroll
  for (int k = 0; k < 32; k += 8)
    tile[ty + k][tx] = X[(long long)(r0 + ty + k) * NN + c0 + tx];
  __syncthreads();
  #pragma unroll
  for (int k = 0; k < 32; k += 8)
    XT[(long long)(c0 + ty + k) * DD + r0 + tx] = f2bf(tile[tx][ty + k]);
}

// ---------------- TDM-staged bf16 WMMA GEMM (linear A addressing) ----------------
// C[M][Ncols] = A[M][K] * B^T, B given as [Ncols][K]. Tiles staged via Tensor Data Mover
// into double-buffered LDS; wave 0 drives the TDM, 8 waves consume via ds_load_b128.
// CMODE: 0 = fp32 linear store, 2 = bf16 transposed store (Cb[col*ldc + row])
template<int CMODE, bool RELU, bool HASBIAS>
__global__ __launch_bounds__(256)
void gemm_bf16_tdm(const uint16_t* __restrict__ A, long long aHead, int lda, int Mtot,
                   const uint16_t* __restrict__ B, long long bHead, int ldb, int Ntot,
                   float* __restrict__ Cf, uint16_t* __restrict__ Cb,
                   long long cHead, int ldc,
                   const float* __restrict__ bias, int K)
{
  extern __shared__ uint16_t lds16[];
  const int lane = threadIdx.x & 31;
  const int wave = threadIdx.x >> 5;
  const int head = blockIdx.z;
  const int mblk = blockIdx.x * 128;
  const int nblk = blockIdx.y * 128;

  const int wm = (wave & 3) * 32;     // 4 waves along M
  const int wn = (wave >> 2) * 64;    // 2 waves along N
  const int half = lane >> 4;
  const int sub  = lane & 15;

  const v8f vz = {0.f,0.f,0.f,0.f,0.f,0.f,0.f,0.f};
  v8f acc[2][4];
  #pragma unroll
  for (int i = 0; i < 2; ++i)
    #pragma unroll
    for (int j = 0; j < 4; ++j) acc[i][j] = vz;

  const uint64_t gA = (uint64_t)A + 2ull * ((uint64_t)head * aHead + (uint64_t)mblk * lda);
  const uint64_t gB = (uint64_t)B + 2ull * ((uint64_t)head * bHead + (uint64_t)nblk * ldb);

  if (wave == 0) {   // prologue: stage chunk 0 into buffer 0
    tdm_issue(gA, 0u,                     (uint32_t)K, (uint32_t)Mtot, (uint32_t)lda);
    tdm_issue(gB, (uint32_t)(CHUNKE * 2), (uint32_t)K, (uint32_t)Ntot, (uint32_t)ldb);
  }

  for (int kc = 0; kc < K; kc += KC) {
    const int buf = (kc / KC) & 1;
    if (wave == 0) {
      if (kc + KC < K) {
        const int nbuf = buf ^ 1;
        tdm_issue(gA + 2ull * (kc + KC), (uint32_t)((nbuf * 2 * CHUNKE) * 2),
                  (uint32_t)K, (uint32_t)Mtot, (uint32_t)lda);
        tdm_issue(gB + 2ull * (kc + KC), (uint32_t)((nbuf * 2 * CHUNKE + CHUNKE) * 2),
                  (uint32_t)K, (uint32_t)Ntot, (uint32_t)ldb);
        __builtin_amdgcn_s_wait_tensorcnt((short)2);  // current chunk's 2 ops done
      } else {
        __builtin_amdgcn_s_wait_tensorcnt((short)0);
      }
    }
    __syncthreads();   // chunk `buf` visible to all waves

    const uint16_t* lA = lds16 + buf * 2 * CHUNKE;
    const uint16_t* lB = lds16 + buf * 2 * CHUNKE + CHUNKE;
    #pragma unroll
    for (int k0 = 0; k0 < KC; k0 += 32) {
      v16bf afr[2];
      #pragma unroll
      for (int i = 0; i < 2; ++i)
        afr[i] = load_frag_a(lA + (wm + 16 * i + sub) * LROW + k0 + half * 8);
      #pragma unroll
      for (int j = 0; j < 4; ++j) {
        v16bf bfr = load_frag_b(lB + (wn + 16 * j + sub) * LROW + k0 + half * 8);
        acc[0][j] = wmma_bf16(afr[0], bfr, acc[0][j]);
        acc[1][j] = wmma_bf16(afr[1], bfr, acc[1][j]);
      }
    }
    __syncthreads();   // done with `buf` before TDM overwrites it two chunks later
  }

  #pragma unroll
  for (int i = 0; i < 2; ++i) {
    #pragma unroll
    for (int r = 0; r < 8; ++r) {
      int grow = mblk + wm + 16 * i + half * 8 + r;
      float bv = HASBIAS ? bias[grow] : 0.0f;
      #pragma unroll
      for (int j = 0; j < 4; ++j) {
        int gcol = nblk + wn + 16 * j + sub;
        float val = acc[i][j][r] + bv;
        if (RELU) val = fmaxf(val, 0.0f);
        if (CMODE == 0) {
          Cf[(long long)head * cHead + (long long)grow * ldc + gcol] = val;
        } else {
          Cb[(long long)gcol * ldc + grow] = f2bf(val);
        }
      }
    }
  }
}

// ---------------- direct-from-L2 bf16 WMMA GEMM (for the CSL-addressed PV GEMM) ----------
// AMODE: 1 = CSL ((row>>6)*16384 + (row&63)*256); CMODE: 1 = fp32 CSL scatter
template<int AMODE, int CMODE, bool RELU, bool HASBIAS>
__global__ __launch_bounds__(256)
void gemm_bf16(const uint16_t* __restrict__ A, long long aHead, int lda,
               const uint16_t* __restrict__ B, long long bHead, int ldb,
               float* __restrict__ Cf, uint16_t* __restrict__ Cb,
               long long cHead, int ldc,
               const float* __restrict__ bias, int K)
{
  const int lane = threadIdx.x & 31;
  const int wave = threadIdx.x >> 5;
  const int head = blockIdx.z;
  A += (long long)head * aHead;
  B += (long long)head * bHead;

  const int m0 = blockIdx.x * 128 + (wave & 3) * 32;
  const int n0 = blockIdx.y * 128 + (wave >> 2) * 64;

  const v8f vz = {0.f,0.f,0.f,0.f,0.f,0.f,0.f,0.f};
  v8f acc[2][4];
  #pragma unroll
  for (int i = 0; i < 2; ++i)
    #pragma unroll
    for (int j = 0; j < 4; ++j) acc[i][j] = vz;

  const int half = lane >> 4;
  const int sub  = lane & 15;

  for (int k0 = 0; k0 < K; k0 += 32) {
    v16bf afr[2];
    #pragma unroll
    for (int i = 0; i < 2; ++i) {
      int row = m0 + 16 * i + sub;
      long long roff = (AMODE == 0)
          ? (long long)row * lda
          : ((long long)(row >> 6) * 16384 + (long long)(row & 63) * 256);
      const uint16_t* p = A + roff + k0 + half * 8;
      __builtin_prefetch(p + 32, 0, 1);
      afr[i] = load_frag_a(p);
    }
    #pragma unroll
    for (int j = 0; j < 4; ++j) {
      int col = n0 + 16 * j + sub;
      const uint16_t* p = B + (long long)col * ldb + k0 + half * 8;
      __builtin_prefetch(p + 32, 0, 1);
      v16bf bfr = load_frag_b(p);
      acc[0][j] = wmma_bf16(afr[0], bfr, acc[0][j]);
      acc[1][j] = wmma_bf16(afr[1], bfr, acc[1][j]);
    }
  }

  #pragma unroll
  for (int i = 0; i < 2; ++i) {
    #pragma unroll
    for (int r = 0; r < 8; ++r) {
      int grow = m0 + 16 * i + half * 8 + r;
      float bv = HASBIAS ? bias[grow] : 0.0f;
      #pragma unroll
      for (int j = 0; j < 4; ++j) {
        int gcol = n0 + 16 * j + sub;
        float val = acc[i][j][r] + bv;
        if (RELU) val = fmaxf(val, 0.0f);
        if (CMODE == 0) {
          Cf[(long long)head * cHead + (long long)grow * ldc + gcol] = val;
        } else if (CMODE == 1) {
          Cf[(long long)head * cHead + (long long)(grow >> 6) * 16384 +
             (long long)(grow & 63) * 256 + gcol] = val;
        } else {
          Cb[(long long)gcol * ldc + grow] = f2bf(val);
        }
      }
    }
  }
}

// ---------------- qkv permute for row attention ----------------
__global__ __launch_bounds__(256)
void permute_qkv(const float* __restrict__ qkv, uint16_t* __restrict__ qT,
                 uint16_t* __restrict__ kT, uint16_t* __restrict__ vb) {
  long long idx = (long long)blockIdx.x * 256 + threadIdx.x;
  int d = (int)(idx >> 14);
  int n = (int)(idx & 16383);
  uint16_t bv = f2bf(qkv[idx]);
  if (d < 512) {
    int h = d >> 3, c = d & 7, s = n >> 8, i = n & 255;
    qT[(long long)h * 131072 + (long long)i * 512 + c * 64 + s] = bv;
  } else if (d < 1024) {
    int dd = d - 512;
    int h = dd >> 3, c = dd & 7, s = n >> 8, j = n & 255;
    kT[(long long)h * 131072 + (long long)j * 512 + c * 64 + s] = bv;
  } else {
    vb[(long long)(d - 1024) * NN + n] = bv;
  }
}

// ---------------- row softmax (one wave per (h,i) row of 256) ----------------
__global__ __launch_bounds__(256)
void row_softmax(const float* __restrict__ sc, uint16_t* __restrict__ attnb) {
  int row  = (blockIdx.x * 256 + threadIdx.x) >> 5;
  int lane = threadIdx.x & 31;
  const float* p = sc + (long long)row * 256;
  float v[8]; float mx = -3.4e38f;
  #pragma unroll
  for (int t = 0; t < 8; ++t) { v[t] = p[lane + 32 * t]; mx = fmaxf(mx, v[t]); }
  #pragma unroll
  for (int o = 16; o > 0; o >>= 1) mx = fmaxf(mx, __shfl_xor(mx, o, 32));
  float sm = 0.f;
  #pragma unroll
  for (int t = 0; t < 8; ++t) { v[t] = __expf(v[t] - mx); sm += v[t]; }
  #pragma unroll
  for (int o = 16; o > 0; o >>= 1) sm += __shfl_xor(sm, o, 32);
  float inv = 1.0f / sm;
  uint16_t* outp = attnb + (long long)row * 256;
  #pragma unroll
  for (int t = 0; t < 8; ++t) outp[lane + 32 * t] = f2bf(v[t] * inv);
}

// ---------------- fused column attention (K = C = 8, online softmax) ----------------
__global__ __launch_bounds__(256)
void col_attn(const float* __restrict__ qkv2, float* __restrict__ colout) {
  int wid  = (blockIdx.x * 256 + threadIdx.x) >> 5;
  int lane = threadIdx.x & 31;
  int h = wid >> 8;
  int l = wid & 255;
  const float* qb = qkv2 + (long long)(h * 8) * NN + l;
  const float* kb = qkv2 + (long long)(512 + h * 8) * NN + l;
  const float* vb = qkv2 + (long long)(1024 + h * 8) * NN + l;
  #pragma unroll
  for (int rep = 0; rep < 2; ++rep) {
    int i = lane + rep * 32;
    float qv[8];
    #pragma unroll
    for (int c = 0; c < 8; ++c) qv[c] = qb[(long long)c * NN + i * 256];
    float mx = -3.4e38f, sm = 0.f;
    for (int j = 0; j < 64; ++j) {
      float s = 0.f;
      #pragma unroll
      for (int c = 0; c < 8; ++c) s += qv[c] * kb[(long long)c * NN + j * 256];
      float nm = fmaxf(mx, s);
      sm = sm * __expf(mx - nm) + __expf(s - nm);
      mx = nm;
    }
    float inv = 1.0f / sm;
    float acc[8];
    #pragma unroll
    for (int c = 0; c < 8; ++c) acc[c] = 0.f;
    for (int j = 0; j < 64; ++j) {
      float s = 0.f;
      #pragma unroll
      for (int c = 0; c < 8; ++c) s += qv[c] * kb[(long long)c * NN + j * 256];
      float w = __expf(s - mx) * inv;
      #pragma unroll
      for (int c = 0; c < 8; ++c) acc[c] += w * vb[(long long)c * NN + j * 256];
    }
    #pragma unroll
    for (int c = 0; c < 8; ++c)
      colout[(long long)(h * 8 + c) * NN + i * 256 + l] = acc[c];
  }
}

// ---------------- channel LayerNorms (thread per column n) ----------------
__global__ __launch_bounds__(256)
void ln_add(const float* __restrict__ a, const float* __restrict__ b,
            const float* __restrict__ w, const float* __restrict__ bb,
            float* __restrict__ outf, uint16_t* __restrict__ outT) {
  int n = blockIdx.x * 256 + threadIdx.x;
  float s = 0.f, s2 = 0.f;
  for (int d = 0; d < DD; ++d) {
    float t = a[(long long)d * NN + n] + b[(long long)d * NN + n];
    s += t; s2 += t * t;
  }
  float mean = s * (1.0f / DD);
  float var  = s2 * (1.0f / DD) - mean * mean;
  float rs   = rsqrtf(var + EPSV);
  for (int d = 0; d < DD; ++d) {
    float t = a[(long long)d * NN + n] + b[(long long)d * NN + n];
    float y = (t - mean) * rs * w[d] + bb[d];
    if (outf) outf[(long long)d * NN + n] = y;
    if (outT) outT[(long long)n * DD + d] = f2bf(y);
  }
}

__global__ __launch_bounds__(256)
void ln_double(const float* __restrict__ a, const float* __restrict__ b,
               const float* __restrict__ w2, const float* __restrict__ b2,
               const float* __restrict__ w1, const float* __restrict__ b1,
               float* __restrict__ hout, uint16_t* __restrict__ hT) {
  int n = blockIdx.x * 256 + threadIdx.x;
  float s = 0.f, s2 = 0.f;
  for (int d = 0; d < DD; ++d) {
    float t = a[(long long)d * NN + n] + b[(long long)d * NN + n];
    s += t; s2 += t * t;
  }
  float m1 = s * (1.0f / DD);
  float v1 = s2 * (1.0f / DD) - m1 * m1;
  float r1 = rsqrtf(v1 + EPSV);
  float sy = 0.f, sy2 = 0.f;
  for (int d = 0; d < DD; ++d) {
    float t = a[(long long)d * NN + n] + b[(long long)d * NN + n];
    float y = (t - m1) * r1 * w2[d] + b2[d];
    sy += y; sy2 += y * y;
  }
  float m2 = sy * (1.0f / DD);
  float v2 = sy2 * (1.0f / DD) - m2 * m2;
  float r2 = rsqrtf(v2 + EPSV);
  for (int d = 0; d < DD; ++d) {
    float t = a[(long long)d * NN + n] + b[(long long)d * NN + n];
    float y = (t - m1) * r1 * w2[d] + b2[d];
    float z = (y - m2) * r2 * w1[d] + b1[d];
    hout[(long long)d * NN + n] = z;
    hT[(long long)n * DD + d] = f2bf(z);
  }
}

// ---------------- host side ----------------
extern "C" void kernel_launch(void* const* d_in, const int* in_sizes, int n_in,
                              void* d_out, int out_size, void* d_ws, size_t ws_size,
                              hipStream_t stream) {
  (void)in_sizes; (void)n_in; (void)out_size; (void)ws_size;
  const float* x     = (const float*)d_in[0];
  const float* w_row = (const float*)d_in[1];
  const float* b_row = (const float*)d_in[2];
  const float* w_col = (const float*)d_in[3];
  const float* b_col = (const float*)d_in[4];
  const float* w_an1 = (const float*)d_in[5];
  const float* b_an1 = (const float*)d_in[6];
  const float* w_an2 = (const float*)d_in[7];
  const float* b_an2 = (const float*)d_in[8];
  const float* w_ff1 = (const float*)d_in[9];
  const float* b_ff1 = (const float*)d_in[10];
  const float* w_ff2 = (const float*)d_in[11];
  const float* b_ff2 = (const float*)d_in[12];
  const float* w_ln1 = (const float*)d_in[13];
  const float* b_ln1 = (const float*)d_in[14];
  const float* w_ln2 = (const float*)d_in[15];
  const float* b_ln2 = (const float*)d_in[16];

  char* ws = (char*)d_ws;
  size_t off = 0;
  auto alloc = [&](size_t bytes) { size_t o = off; off += (bytes + 255) & ~(size_t)255; return o; };

  uint16_t* xbT  = (uint16_t*)(ws + alloc((size_t)NN * DD * 2));
  uint16_t* wrow = (uint16_t*)(ws + alloc((size_t)1536 * DD * 2));
  uint16_t* wcol = (uint16_t*)(ws + alloc((size_t)1536 * DD * 2));
  uint16_t* wff1 = (uint16_t*)(ws + alloc((size_t)FFD * DD * 2));
  uint16_t* wff2 = (uint16_t*)(ws + alloc((size_t)DD * FFD * 2));
  float*    qkv  = (float*)   (ws + alloc((size_t)1536 * NN * 4));
  uint16_t* qT   = (uint16_t*)(ws + alloc((size_t)HH * 256 * 512 * 2));
  uint16_t* kT   = (uint16_t*)(ws + alloc((size_t)HH * 256 * 512 * 2));
  uint16_t* vb   = (uint16_t*)(ws + alloc((size_t)DD * NN * 2));
  float*    scor = (float*)   (ws + alloc((size_t)HH * 256 * 256 * 4));
  uint16_t* attn = (uint16_t*)(ws + alloc((size_t)HH * 256 * 256 * 2));
  float*    rowo = (float*)   (ws + alloc((size_t)DD * NN * 4));
  float*    out1 = (float*)   (ws + alloc((size_t)DD * NN * 4));
  uint16_t* x1T  = (uint16_t*)(ws + alloc((size_t)NN * DD * 2));
  // aliases (lifetimes disjoint in the stream-ordered sequence)
  float*    colo = (float*)qT;
  float*    hbuf = rowo;
  uint16_t* hT   = vb;
  float*    ffb  = out1;
  uint16_t* fT   = (uint16_t*)qkv;
  // ~286 MB total workspace

  // 1) casts
  cast_f32_bf16<<<(1536*512+255)/256, 256, 0, stream>>>(w_row, wrow, 1536*512);
  cast_f32_bf16<<<(1536*512+255)/256, 256, 0, stream>>>(w_col, wcol, 1536*512);
  cast_f32_bf16<<<(FFD*DD+255)/256,   256, 0, stream>>>(w_ff1, wff1, FFD*DD);
  cast_f32_bf16<<<(DD*FFD+255)/256,   256, 0, stream>>>(w_ff2, wff2, DD*FFD);
  transpose_cast_x<<<dim3(NN/32, DD/32), 256, 0, stream>>>(x, xbT);

  // 2) row qkv (TDM-staged): [1536,16384,K=512]
  gemm_bf16_tdm<0,false,true><<<dim3(12, 128, 1), 256, SMEM_BYTES, stream>>>(
      wrow, 0, DD, 1536, xbT, 0, DD, NN, qkv, nullptr, 0, NN, b_row, DD);

  // 3) permute to per-head layouts
  permute_qkv<<<(1536*NN)/256, 256, 0, stream>>>(qkv, qT, kT, vb);

  // 4) row scores per head (TDM-staged): [256,256,K=512]
  gemm_bf16_tdm<0,false,false><<<dim3(2, 2, HH), 256, SMEM_BYTES, stream>>>(
      qT, 131072, 512, 256, kT, 131072, 512, 256, scor, nullptr, 65536, 256, nullptr, 512);

  // 5) softmax over j -> bf16 probs
  row_softmax<<<(HH*256)/8, 256, 0, stream>>>(scor, attn);

  // 6) PV per head (direct path; CSL A-addressing): [512,256,K=256]
  gemm_bf16<1,1,false,false><<<dim3(4, 2, HH), 256, 0, stream>>>(
      vb, 131072, 0, attn, 65536, 256, rowo, nullptr, 131072, 0, nullptr, 256);

  // 7) out1 = LN(x + rowout, an1); + bf16 transposed copy
  ln_add<<<NN/256, 256, 0, stream>>>(x, rowo, w_an1, b_an1, out1, x1T);

  // 8) col qkv (TDM-staged)
  gemm_bf16_tdm<0,false,true><<<dim3(12, 128, 1), 256, SMEM_BYTES, stream>>>(
      wcol, 0, DD, 1536, x1T, 0, DD, NN, qkv, nullptr, 0, NN, b_col, DD);

  // 9) fused column attention (K=8)
  col_attn<<<(HH*LLEN)/8, 256, 0, stream>>>(qkv, colo);

  // 10) h = LN(LN(out1 + colout, an2), ln1); + bf16 transposed copy
  ln_double<<<NN/256, 256, 0, stream>>>(out1, colo, w_an2, b_an2, w_ln1, b_ln1, hbuf, hT);

  // 11) FF1 (TDM-staged) with fused bias+ReLU, transposed bf16 store [N][F]
  gemm_bf16_tdm<2,true,true><<<dim3(FFD/128, 128, 1), 256, SMEM_BYTES, stream>>>(
      wff1, 0, DD, FFD, hT, 0, DD, NN, nullptr, fT, 0, FFD, b_ff1, DD);

  // 12) FF2 (TDM-staged): [512,16384,K=2048]
  gemm_bf16_tdm<0,false,true><<<dim3(4, 128, 1), 256, SMEM_BYTES, stream>>>(
      wff2, 0, FFD, DD, fT, 0, FFD, NN, ffb, nullptr, 0, NN, b_ff2, FFD);

  // 13) final: out = LN(h + ff, ln2)
  ln_add<<<NN/256, 256, 0, stream>>>(hbuf, ffb, w_ln2, b_ln2, (float*)d_out, nullptr);
}
